// BiSSDLayer_24197845746208
// MI455X (gfx1250) — matrix operand order repaired and checked by reading, not compiled, not run on hardware
//
#include <hip/hip_runtime.h>
#include <cstddef>

#define D_MODEL   512
#define D_STATE   64
#define D_CONV    4
#define D_INNER   1024
#define NHEADS    16
#define HEADDIM   64
#define D_IN_PROJ 2192   // 2*D_INNER + 2*D_STATE + NHEADS
#define CONV_DIM  1152   // D_INNER + 2*D_STATE
#define BATCHN    8
#define SEQLEN    1024
#define MROWS     (BATCHN * SEQLEN)   // 8192
#define D_FFN     (4 * D_MODEL)       // 2048

typedef __attribute__((ext_vector_type(16))) __bf16 bf16x16;
typedef __attribute__((ext_vector_type(8)))  float  f32x8;

struct alignas(16) U4 { unsigned int x, y, z, w; };

union Frag {
  bf16x16 v;
  U4 q[2];
};

__device__ __forceinline__ unsigned short f32_to_bf16(float f) {
  unsigned int u = __builtin_bit_cast(unsigned int, f);
  u += 0x7FFFu + ((u >> 16) & 1u);   // round-to-nearest-even
  return (unsigned short)(u >> 16);
}

__device__ __forceinline__ float sigmoidf_(float x) { return 1.f / (1.f + __expf(-x)); }

// ---------------------------------------------------------------------------
// bf16 WMMA GEMM: Out[M,N](f32) = A[M,K](bf16) @ W[N,K](bf16)^T
// Branch-free K-loop: tile counts are template params. Per wave: MT*16 rows x
// NT*16 cols -> MT*NT independent accumulator chains, MT+NT fragment loads
// shared by MT*NT WMMAs per 32-wide K step.
// block = 128 threads (4 waves stacked on M); block tile = (64*MT) x (16*NT)
// ---------------------------------------------------------------------------
template<int MT, int NT>
__global__ __launch_bounds__(128)
void gemm_wmma(const unsigned short* __restrict__ A,
               const unsigned short* __restrict__ W,
               float* __restrict__ Out,
               int M, int N, int K, int n0base)
{
  const int lane = threadIdx.x & 31;
  const int wave = threadIdx.x >> 5;
  const int m0 = blockIdx.x * (64 * MT) + wave * (16 * MT);
  const int n0 = n0base + blockIdx.y * (16 * NT);

  // A fragment (16x32 bf16): lane m = lane&15; K halves at o and o+16
  const int ar = lane & 15;
  const int ka = (lane >> 4) * 8;
  // B fragment (32x16 bf16 from row-major W[N,K]): lane n; K halves at o and o+16
  const int bn = lane & 15;
  const int kb = (lane >> 4) * 8;

  f32x8 zero = {0.f,0.f,0.f,0.f,0.f,0.f,0.f,0.f};
  f32x8 acc[MT][NT];
#pragma unroll
  for (int i = 0; i < MT; ++i)
#pragma unroll
    for (int j = 0; j < NT; ++j) acc[i][j] = zero;

  for (int k0 = 0; k0 < K; k0 += 32) {
    Frag fa[MT];
#pragma unroll
    for (int i = 0; i < MT; ++i) {
      const unsigned short* pa = A + (size_t)(m0 + i * 16 + ar) * K + k0 + ka;
      fa[i].q[0] = *(const U4*)(pa);
      fa[i].q[1] = *(const U4*)(pa + 16);
    }
    Frag fb[NT];
#pragma unroll
    for (int j = 0; j < NT; ++j) {
      const unsigned short* pb = W + (size_t)(n0 + j * 16 + bn) * K + k0 + kb;
      fb[j].q[0] = *(const U4*)(pb);
      fb[j].q[1] = *(const U4*)(pb + 16);
    }
    if (k0 + 32 < K)
      __builtin_prefetch((const void*)(A + (size_t)(m0 + ar) * K + k0 + 32 + ka), 0, 1);

#pragma unroll
    for (int i = 0; i < MT; ++i)
#pragma unroll
      for (int j = 0; j < NT; ++j)
        acc[i][j] = __builtin_amdgcn_wmma_f32_16x16x32_bf16(
            false, fa[i].v, false, fb[j].v, (short)0, acc[i][j], false, false);
  }

  // D layout: lane%16 = N col; VGPR r = M row (upper half-lanes: +8)
  const int mB = (lane >> 4) * 8;
  const int nc = lane & 15;
#pragma unroll
  for (int i = 0; i < MT; ++i)
#pragma unroll
    for (int j = 0; j < NT; ++j) {
      float* po = Out + (size_t)(m0 + i * 16 + mB) * N + n0 + j * 16 + nc;
#pragma unroll
      for (int r = 0; r < 8; ++r) po[(size_t)r * N] = acc[i][j][r];
    }
}

// ---------------------------------------------------------------------------
// elementwise / small kernels
// ---------------------------------------------------------------------------
__global__ void cvt_f32_bf16(const float* __restrict__ in,
                             unsigned short* __restrict__ out, int n)
{
  int i = blockIdx.x * 256 + threadIdx.x;
  if (i < n) out[i] = f32_to_bf16(in[i]);
}

// item_emb -> bf16 (fwd) + seq-flipped bf16 (bwd input)
__global__ void prep_inputs(const float* __restrict__ emb,
                            unsigned short* __restrict__ embBf,
                            unsigned short* __restrict__ flipBf)
{
  size_t idx = (size_t)blockIdx.x * 256 + threadIdx.x;
  if (idx >= (size_t)MROWS * D_MODEL) return;
  int d = (int)(idx % D_MODEL);
  size_t bs = idx / D_MODEL;
  int s = (int)(bs % SEQLEN);
  int b = (int)(bs / SEQLEN);
  float v = emb[idx];
  embBf[idx] = f32_to_bf16(v);
  size_t fidx = ((size_t)b * SEQLEN + (SEQLEN - 1 - s)) * D_MODEL + d;
  flipBf[fidx] = f32_to_bf16(v);
}

__global__ void dt_softplus(const float* __restrict__ zx,
                            const float* __restrict__ dt_bias,
                            float* __restrict__ dt)
{
  int idx = blockIdx.x * 256 + threadIdx.x;
  if (idx >= MROWS * NHEADS) return;
  int h = idx & (NHEADS - 1);
  size_t r = (size_t)(idx >> 4);
  float v = zx[r * D_IN_PROJ + D_INNER + CONV_DIM + h] + dt_bias[h];
  dt[idx] = (v > 20.f) ? v : log1pf(__expf(v));
}

// causal depthwise conv (width 4) + bias + silu over xBC slice of zxbcdt
__global__ void conv_silu(const float* __restrict__ zx,
                          const float* __restrict__ cw,
                          const float* __restrict__ cb,
                          float* __restrict__ xc)
{
  size_t idx = (size_t)blockIdx.x * 256 + threadIdx.x;
  if (idx >= (size_t)MROWS * CONV_DIM) return;
  int c = (int)(idx % CONV_DIM);
  size_t bs = idx / CONV_DIM;
  int s = (int)(bs % SEQLEN);
  size_t b = bs / SEQLEN;
  float acc = cb[c];
#pragma unroll
  for (int j = 0; j < D_CONV; ++j) {
    int t = s - (D_CONV - 1) + j;
    if (t >= 0)
      acc += zx[((size_t)b * SEQLEN + t) * D_IN_PROJ + D_INNER + c] * cw[c * D_CONV + j];
  }
  xc[idx] = acc * sigmoidf_(acc);
}

// SSD sequential scan: one block per (batch, head); 64 threads (p = headdim)
#define SCH 32
__global__ __launch_bounds__(64)
void ssd_scan(const float* __restrict__ xc,   // (B,S,CONV_DIM)
              const float* __restrict__ dt,   // (B,S,NHEADS)
              const float* __restrict__ A_log,
              const float* __restrict__ Dp,
              float* __restrict__ y)          // (B,S,D_INNER)
{
  const int b = blockIdx.x / NHEADS;
  const int h = blockIdx.x % NHEADS;
  const int p = threadIdx.x;                  // 0..63

  __shared__ float Bs[SCH][D_STATE];
  __shared__ float Cs[SCH][D_STATE];
  __shared__ float Xs[SCH][HEADDIM];
  __shared__ float dts[SCH];
  __shared__ float dAs[SCH];

  const float Ah = -__expf(A_log[h]);
  const float Dh = Dp[h];

  float hs[D_STATE];
#pragma unroll
  for (int n = 0; n < D_STATE; ++n) hs[n] = 0.f;

  for (int t0 = 0; t0 < SEQLEN; t0 += SCH) {
    for (int i = threadIdx.x; i < SCH * 64; i += 64) {
      int tt = i >> 6, n = i & 63;
      const float* row = xc + ((size_t)b * SEQLEN + t0 + tt) * CONV_DIM;
      Bs[tt][n] = row[D_INNER + n];
      Cs[tt][n] = row[D_INNER + D_STATE + n];
      Xs[tt][n] = row[h * HEADDIM + n];
    }
    for (int i = threadIdx.x; i < SCH; i += 64) {
      float dv = dt[((size_t)b * SEQLEN + t0 + i) * NHEADS + h];
      dts[i] = dv;
      dAs[i] = __expf(dv * Ah);
    }
    __syncthreads();

    for (int tt = 0; tt < SCH; ++tt) {
      const float dAv = dAs[tt];
      const float dtv = dts[tt];
      const float xv  = Xs[tt][p];
      const float dtx = dtv * xv;
      float acc = 0.f;
#pragma unroll
      for (int n = 0; n < D_STATE; ++n) {
        hs[n] = fmaf(hs[n], dAv, dtx * Bs[tt][n]);
        acc = fmaf(hs[n], Cs[tt][n], acc);
      }
      y[((size_t)b * SEQLEN + t0 + tt) * D_INNER + h * HEADDIM + p] = fmaf(xv, Dh, acc);
    }
    __syncthreads();
  }
}

// g = y * silu(z); RMSNorm * norm_w; -> bf16 (out_proj GEMM input)
__global__ __launch_bounds__(256)
void gate_rmsnorm(const float* __restrict__ y,
                  const float* __restrict__ zx,
                  const float* __restrict__ nw,
                  unsigned short* __restrict__ gBf)
{
  const int row = blockIdx.x;
  const float* yr = y + (size_t)row * D_INNER;
  const float* zr = zx + (size_t)row * D_IN_PROJ;   // z = first D_INNER cols
  __shared__ float red[256];
  float g[4]; float ss = 0.f;
#pragma unroll
  for (int i = 0; i < 4; ++i) {
    int c = threadIdx.x + i * 256;
    float z = zr[c];
    float gv = yr[c] * (z * sigmoidf_(z));
    g[i] = gv; ss += gv * gv;
  }
  red[threadIdx.x] = ss; __syncthreads();
  for (int o = 128; o > 0; o >>= 1) {
    if (threadIdx.x < o) red[threadIdx.x] += red[threadIdx.x + o];
    __syncthreads();
  }
  float rms = rsqrtf(red[0] * (1.f / D_INNER) + 1e-5f);
#pragma unroll
  for (int i = 0; i < 4; ++i) {
    int c = threadIdx.x + i * 256;
    gBf[(size_t)row * D_INNER + c] = f32_to_bf16(g[i] * rms * nw[c]);
  }
}

// h = fwd + BETA*bwd + emb; LayerNorm -> h_f32 (residual) + h_bf16 (FFN input)
__global__ __launch_bounds__(256)
void combine_ln(const float* __restrict__ fwd,
                const float* __restrict__ bwd,
                const float* __restrict__ emb,
                const float* __restrict__ lnw,
                const float* __restrict__ lnb,
                float* __restrict__ hRes,
                unsigned short* __restrict__ hBf)
{
  const int row = blockIdx.x;
  __shared__ float r1[256], r2[256];
  float v[2]; float s = 0.f, s2 = 0.f;
#pragma unroll
  for (int i = 0; i < 2; ++i) {
    size_t c = (size_t)row * D_MODEL + threadIdx.x + i * 256;
    float x = fwd[c] + 0.5f * bwd[c] + emb[c];
    v[i] = x; s += x; s2 += x * x;
  }
  r1[threadIdx.x] = s; r2[threadIdx.x] = s2; __syncthreads();
  for (int o = 128; o > 0; o >>= 1) {
    if (threadIdx.x < o) { r1[threadIdx.x] += r1[threadIdx.x + o]; r2[threadIdx.x] += r2[threadIdx.x + o]; }
    __syncthreads();
  }
  float mu  = r1[0] * (1.f / D_MODEL);
  float var = r2[0] * (1.f / D_MODEL) - mu * mu;
  float inv = rsqrtf(var + 1e-12f);
#pragma unroll
  for (int i = 0; i < 2; ++i) {
    int c = threadIdx.x + i * 256;
    float o = (v[i] - mu) * inv * lnw[c] + lnb[c];
    hRes[(size_t)row * D_MODEL + c] = o;
    hBf[(size_t)row * D_MODEL + c] = f32_to_bf16(o);
  }
}

__global__ void bias_gelu_bf16(const float* __restrict__ h1,
                               const float* __restrict__ b1,
                               unsigned short* __restrict__ h1Bf)
{
  size_t idx = (size_t)blockIdx.x * 256 + threadIdx.x;
  if (idx >= (size_t)MROWS * D_FFN) return;
  int c = (int)(idx % D_FFN);
  float v = h1[idx] + b1[c];
  float g = 0.5f * v * (1.f + erff(v * 0.70710678118654752f));
  h1Bf[idx] = f32_to_bf16(g);
}

__global__ __launch_bounds__(256)
void final_ln(const float* __restrict__ f2,
              const float* __restrict__ b2,
              const float* __restrict__ hRes,
              const float* __restrict__ lnw,
              const float* __restrict__ lnb,
              float* __restrict__ out)
{
  const int row = blockIdx.x;
  __shared__ float r1[256], r2[256];
  float v[2]; float s = 0.f, s2 = 0.f;
#pragma unroll
  for (int i = 0; i < 2; ++i) {
    int c = threadIdx.x + i * 256;
    float x = f2[(size_t)row * D_MODEL + c] + b2[c] + hRes[(size_t)row * D_MODEL + c];
    v[i] = x; s += x; s2 += x * x;
  }
  r1[threadIdx.x] = s; r2[threadIdx.x] = s2; __syncthreads();
  for (int o = 128; o > 0; o >>= 1) {
    if (threadIdx.x < o) { r1[threadIdx.x] += r1[threadIdx.x + o]; r2[threadIdx.x] += r2[threadIdx.x + o]; }
    __syncthreads();
  }
  float mu  = r1[0] * (1.f / D_MODEL);
  float var = r2[0] * (1.f / D_MODEL) - mu * mu;
  float inv = rsqrtf(var + 1e-12f);
#pragma unroll
  for (int i = 0; i < 2; ++i) {
    int c = threadIdx.x + i * 256;
    out[(size_t)row * D_MODEL + c] = (v[i] - mu) * inv * lnw[c] + lnb[c];
  }
}

// ---------------------------------------------------------------------------
// main GEMM launcher: main region with NT=4, 16-wide tail (N % 64) with NT=1
// ---------------------------------------------------------------------------
static void run_gemm(const unsigned short* A, const unsigned short* W, float* O,
                     int M, int N, int K, hipStream_t s)
{
  const int nMain = (N / 64) * 64;
  if (nMain > 0) {
    dim3 grid(M / 128, nMain / 64);
    gemm_wmma<2, 4><<<grid, 128, 0, s>>>(A, W, O, M, N, K, 0);
  }
  const int tail = N - nMain;   // multiple of 16 for all shapes here
  if (tail > 0) {
    dim3 grid(M / 128, tail / 16);
    gemm_wmma<2, 1><<<grid, 128, 0, s>>>(A, W, O, M, N, K, nMain);
  }
}

static inline int blks(size_t n) { return (int)((n + 255) / 256); }

extern "C" void kernel_launch(void* const* d_in, const int* in_sizes, int n_in,
                              void* d_out, int out_size, void* d_ws, size_t ws_size,
                              hipStream_t stream)
{
  const float* item_emb   = (const float*)d_in[0];
  // d_in[1] item_idx, d_in[2] flip_index: unused (flip is static arange reversal)
  const float* in_proj_w  = (const float*)d_in[3];
  const float* conv_w     = (const float*)d_in[4];
  const float* conv_b     = (const float*)d_in[5];
  const float* dt_bias    = (const float*)d_in[6];
  const float* A_log      = (const float*)d_in[7];
  const float* Dp         = (const float*)d_in[8];
  const float* norm_w     = (const float*)d_in[9];
  const float* out_proj_w = (const float*)d_in[10];
  const float* ln_w       = (const float*)d_in[11];
  const float* ln_b       = (const float*)d_in[12];
  const float* ffn_w1     = (const float*)d_in[13];
  const float* ffn_b1     = (const float*)d_in[14];
  const float* ffn_w2     = (const float*)d_in[15];
  const float* ffn_b2     = (const float*)d_in[16];
  const float* ffn_ln_w   = (const float*)d_in[17];
  const float* ffn_ln_b   = (const float*)d_in[18];
  float* out = (float*)d_out;

  char* ws = (char*)d_ws;
  size_t off = 0;
  auto alloc = [&](size_t bytes) -> void* {
    void* p = ws + off;
    off += (bytes + 255) & ~(size_t)255;
    return p;
  };

  unsigned short* wIn    = (unsigned short*)alloc((size_t)D_IN_PROJ * D_MODEL * 2);
  unsigned short* wOut   = (unsigned short*)alloc((size_t)D_MODEL * D_INNER * 2);
  unsigned short* wF1    = (unsigned short*)alloc((size_t)D_FFN * D_MODEL * 2);
  unsigned short* wF2    = (unsigned short*)alloc((size_t)D_MODEL * D_FFN * 2);
  unsigned short* embBf  = (unsigned short*)alloc((size_t)MROWS * D_MODEL * 2);
  unsigned short* flipBf = (unsigned short*)alloc((size_t)MROWS * D_MODEL * 2);
  float* zx   = (float*)alloc((size_t)MROWS * D_IN_PROJ * 4);   // reused as h1 (FFN hidden, f32)
  float* xc   = (float*)alloc((size_t)MROWS * CONV_DIM * 4);    // reused as h1 bf16
  float* dtb  = (float*)alloc((size_t)MROWS * NHEADS * 4);
  float* ybuf = (float*)alloc((size_t)MROWS * D_INNER * 4);     // reused as f2 (f32)
  unsigned short* gBf = (unsigned short*)alloc((size_t)MROWS * D_INNER * 2);
  float* moF  = (float*)alloc((size_t)MROWS * D_MODEL * 4);
  float* moB  = (float*)alloc((size_t)MROWS * D_MODEL * 4);
  float* hRes = (float*)alloc((size_t)MROWS * D_MODEL * 4);
  unsigned short* hBf = (unsigned short*)alloc((size_t)MROWS * D_MODEL * 2);

  // weights -> bf16
  cvt_f32_bf16<<<blks((size_t)D_IN_PROJ * D_MODEL), 256, 0, stream>>>(in_proj_w, wIn, D_IN_PROJ * D_MODEL);
  cvt_f32_bf16<<<blks((size_t)D_MODEL * D_INNER), 256, 0, stream>>>(out_proj_w, wOut, D_MODEL * D_INNER);
  cvt_f32_bf16<<<blks((size_t)D_FFN * D_MODEL), 256, 0, stream>>>(ffn_w1, wF1, D_FFN * D_MODEL);
  cvt_f32_bf16<<<blks((size_t)D_MODEL * D_FFN), 256, 0, stream>>>(ffn_w2, wF2, D_MODEL * D_FFN);
  prep_inputs<<<blks((size_t)MROWS * D_MODEL), 256, 0, stream>>>(item_emb, embBf, flipBf);

  // two Mamba2 passes (fwd + seq-reversed)
  for (int dir = 0; dir < 2; ++dir) {
    const unsigned short* act = (dir == 0) ? embBf : flipBf;
    float* mo = (dir == 0) ? moF : moB;

    run_gemm(act, wIn, zx, MROWS, D_IN_PROJ, D_MODEL, stream);
    dt_softplus<<<blks((size_t)MROWS * NHEADS), 256, 0, stream>>>(zx, dt_bias, dtb);
    conv_silu<<<blks((size_t)MROWS * CONV_DIM), 256, 0, stream>>>(zx, conv_w, conv_b, xc);
    ssd_scan<<<BATCHN * NHEADS, 64, 0, stream>>>(xc, dtb, A_log, Dp, ybuf);
    gate_rmsnorm<<<MROWS, 256, 0, stream>>>(ybuf, zx, norm_w, gBf);
    run_gemm(gBf, wOut, mo, MROWS, D_MODEL, D_INNER, stream);
  }

  // combine + LayerNorm (note: reference adds bwd WITHOUT unflipping)
  combine_ln<<<MROWS, 256, 0, stream>>>(moF, moB, item_emb, ln_w, ln_b, hRes, hBf);

  // FFN (reusing dead buffers)
  float* h1 = zx;                          // 8192x2048 f32 fits in zx region
  unsigned short* h1Bf = (unsigned short*)xc;
  float* f2 = ybuf;

  run_gemm(hBf, wF1, h1, MROWS, D_FFN, D_MODEL, stream);
  bias_gelu_bf16<<<blks((size_t)MROWS * D_FFN), 256, 0, stream>>>(h1, ffn_b1, h1Bf);
  run_gemm(h1Bf, wF2, f2, MROWS, D_MODEL, D_FFN, stream);
  final_ln<<<MROWS, 256, 0, stream>>>(f2, ffn_b2, hRes, ffn_ln_w, ffn_ln_b, out);
}
